// CausalAttNetV1_42674795053751
// MI455X (gfx1250) — matrix operand order repaired and checked by reading, not compiled, not run on hardware
//
#include <hip/hip_runtime.h>
#include <hip/hip_bf16.h>

typedef __attribute__((ext_vector_type(16))) _Float16 v16h;
typedef __attribute__((ext_vector_type(8)))  _Float16 v8h;
typedef __attribute__((ext_vector_type(8)))  float    v8f;

#define Bb 8
#define Nn 256
#define FD 1024
#define Hh 256
#define INV_TEMP 2.0f          // 1 / TEMP
#define KSEL 39321             // int(0.6 * 256 * 256)
#define ROWS (Bb * Nn)         // 2048
#define PCOLS 512              // [pr+b1 | pc]
#define EPB (Nn * Nn)          // 65536 elements per batch
#define OUTR (Bb * EPB)        // 524288 per output tensor

// ---------------------------------------------------------------- conversions
__global__ __launch_bounds__(256)
void cvt_x_kernel(const float* __restrict__ x, _Float16* __restrict__ Xh) {
    int i = (blockIdx.x * 256 + threadIdx.x) * 4;   // 2048 blocks cover 2,097,152
    float4 v = *(const float4*)(x + i);
    Xh[i + 0] = (_Float16)v.x;
    Xh[i + 1] = (_Float16)v.y;
    Xh[i + 2] = (_Float16)v.z;
    Xh[i + 3] = (_Float16)v.w;
}

// W1 is (256, 2048). Build Wh as (512, 1024): row n<256 = W1[n, 0:1024],
// row 256+n = W1[n, 1024:2048]. Row-major so the WMMA B fragment is one
// contiguous 32B load per lane.
__global__ __launch_bounds__(256)
void cvt_w1_kernel(const float* __restrict__ W1, _Float16* __restrict__ Wh) {
    int i = (blockIdx.x * 256 + threadIdx.x) * 4;   // 512 blocks cover 524,288
    int n = i >> 10;
    int k = i & 1023;
    const float* src = W1 + (size_t)(n & 255) * 2048 + (size_t)(n >> 8) * 1024 + k;
    float4 v = *(const float4*)src;
    Wh[i + 0] = (_Float16)v.x;
    Wh[i + 1] = (_Float16)v.y;
    Wh[i + 2] = (_Float16)v.z;
    Wh[i + 3] = (_Float16)v.w;
}

// ---------------------------------------------------------------- WMMA GEMM
// P[2048, 512] = Xh[2048, 1024] @ Wh[512, 1024]^T ; cols 0..255 get +b1.
// 2x2 register blocking: each wave owns a 32x32 output block (four 16x16
// WMMA tiles), reusing each A/B fragment twice -> 2 b128 loads per WMMA.
// 64 Mblocks * 16 Nblocks = 1024 wave-tiles, 8 waves/block -> 128 blocks.
__device__ __forceinline__ v16h load_afrag(const _Float16* __restrict__ arow,
                                           int k0, int hiA) {
    v8h a0 = *(const v8h*)(arow + k0 + hiA);
    v8h a1 = *(const v8h*)(arow + k0 + hiA + 8);
    v16h af;
#pragma unroll
    for (int i = 0; i < 8; i++) { af[i] = a0[i]; af[8 + i] = a1[i]; }
    return af;
}

__global__ __launch_bounds__(256)
void gemm_kernel(const _Float16* __restrict__ Xh, const _Float16* __restrict__ Wh,
                 const float* __restrict__ b1, float* __restrict__ P) {
    int lane = threadIdx.x & 31;
    int wave = threadIdx.x >> 5;
    int tile = blockIdx.x * 8 + wave;     // 0..1023
    int mt = tile >> 4;                   // 0..63
    int nt = tile & 15;                   // 0..15
    int M0 = mt * 32, N0 = nt * 32;
    int l15 = lane & 15;
    int hiA = (lane >> 4) * 16;           // A: K half select
    int hiB = (lane >> 4) * 16;           // B: K half select
    const _Float16* arow0 = Xh + (size_t)(M0 + l15) * FD;
    const _Float16* arow1 = arow0 + (size_t)16 * FD;
    const _Float16* brow0 = Wh + (size_t)(N0 + l15) * FD;
    const _Float16* brow1 = brow0 + (size_t)16 * FD;

    v8f c00 = {}, c01 = {}, c10 = {}, c11 = {};
    for (int k0 = 0; k0 < FD; k0 += 32) {
        __builtin_prefetch(arow0 + k0 + 128, 0, 1);
        __builtin_prefetch(arow1 + k0 + 128, 0, 1);
        __builtin_prefetch(brow0 + k0 + 128, 0, 1);
        __builtin_prefetch(brow1 + k0 + 128, 0, 1);
        v16h af0 = load_afrag(arow0, k0, hiA);
        v16h af1 = load_afrag(arow1, k0, hiA);
        v16h bf0 = *(const v16h*)(brow0 + k0 + hiB);
        v16h bf1 = *(const v16h*)(brow1 + k0 + hiB);
        c00 = __builtin_amdgcn_wmma_f32_16x16x32_f16(false, af0, false, bf0,
                                                     (short)0, c00, false, false);
        c01 = __builtin_amdgcn_wmma_f32_16x16x32_f16(false, af0, false, bf1,
                                                     (short)0, c01, false, false);
        c10 = __builtin_amdgcn_wmma_f32_16x16x32_f16(false, af1, false, bf0,
                                                     (short)0, c10, false, false);
        c11 = __builtin_amdgcn_wmma_f32_16x16x32_f16(false, af1, false, bf1,
                                                     (short)0, c11, false, false);
    }

    int n0 = N0 + l15;
    int n1 = n0 + 16;
    int mrow0 = M0 + (lane >> 4) * 8;
    int mrow1 = mrow0 + 16;
    float bias0 = (n0 < Hh) ? b1[n0] : 0.0f;
    float bias1 = (n1 < Hh) ? b1[n1] : 0.0f;
#pragma unroll
    for (int v = 0; v < 8; v++) {
        P[(size_t)(mrow0 + v) * PCOLS + n0] = c00[v] + bias0;
        P[(size_t)(mrow0 + v) * PCOLS + n1] = c01[v] + bias1;
        P[(size_t)(mrow1 + v) * PCOLS + n0] = c10[v] + bias0;
        P[(size_t)(mrow1 + v) * PCOLS + n1] = c11[v] + bias1;
    }
}

// ---------------------------------------------------------------- fused edge
// edge[b,i,j] = sum_h relu(PRb[b,i,h] + PC[b,j,h]) * W2[h] + b2
// Block = (b, 16-row i-tile); thread = j. 16 i-rows staged in LDS for reuse.
__global__ __launch_bounds__(256)
void edge_kernel(const float* __restrict__ P, const float* __restrict__ W2,
                 const float* __restrict__ b2, float* __restrict__ edge) {
    __shared__ float As[16][257];
    __shared__ float w2s[256];
    int tid = threadIdx.x;
    int b  = blockIdx.x >> 4;
    int it = blockIdx.x & 15;
    int rowBase = b * Nn + it * 16;
#pragma unroll
    for (int ii = 0; ii < 16; ii++)
        As[ii][tid] = P[(size_t)(rowBase + ii) * PCOLS + tid];
    w2s[tid] = W2[tid];
    __syncthreads();

    float acc[16];
#pragma unroll
    for (int ii = 0; ii < 16; ii++) acc[ii] = 0.0f;

    const float* pc = P + (size_t)(b * Nn + tid) * PCOLS + Hh;
    for (int h4 = 0; h4 < Hh; h4 += 4) {
        float4 pv = *(const float4*)(pc + h4);
        float ph[4] = {pv.x, pv.y, pv.z, pv.w};
#pragma unroll
        for (int s = 0; s < 4; s++) {
            int h = h4 + s;
            float w = w2s[h];
#pragma unroll
            for (int ii = 0; ii < 16; ii++)
                acc[ii] += fmaxf(As[ii][h] + ph[s], 0.0f) * w;
        }
    }
    float bb = b2[0];
#pragma unroll
    for (int ii = 0; ii < 16; ii++)
        edge[(size_t)b * EPB + (size_t)(it * 16 + ii) * Nn + tid] = acc[ii] + bb;
}

// ---------------------------------------------------------- per-batch stats
// One block per batch: max -> sum(exp) -> 4-pass radix select (k-th largest)
// -> kept softmax mass. Results: stats[b]=mx, [8+b]=invZ, [16+b]=thr,
// [24+b]=1/(kept_mass + 1e-12).
__device__ __forceinline__ unsigned f2key(float v) {
    unsigned u = __float_as_uint(v);
    return (u & 0x80000000u) ? ~u : (u | 0x80000000u);
}

__global__ __launch_bounds__(256)
void stats_kernel(const float* __restrict__ edge, float* __restrict__ stats) {
    __shared__ float red[256];
    __shared__ unsigned hist[256];
    __shared__ unsigned sh_prefix, sh_remaining;
    __shared__ float sh_mx, sh_invZ, sh_thr;
    int b = blockIdx.x, tid = threadIdx.x;
    const float* s = edge + (size_t)b * EPB;

    // 1) max
    float m = -3.402823466e38f;
    for (int i = tid; i < EPB; i += 256) m = fmaxf(m, s[i]);
    red[tid] = m; __syncthreads();
    for (int w = 128; w > 0; w >>= 1) {
        if (tid < w) red[tid] = fmaxf(red[tid], red[tid + w]);
        __syncthreads();
    }
    if (tid == 0) sh_mx = red[0];
    __syncthreads();
    float mx = sh_mx;

    // 2) softmax denominator
    float acc = 0.0f;
    for (int i = tid; i < EPB; i += 256) acc += __expf((s[i] - mx) * INV_TEMP);
    red[tid] = acc; __syncthreads();
    for (int w = 128; w > 0; w >>= 1) {
        if (tid < w) red[tid] += red[tid + w];
        __syncthreads();
    }
    if (tid == 0) { sh_invZ = 1.0f / red[0]; sh_prefix = 0u; sh_remaining = KSEL; }
    __syncthreads();
    float invZ = sh_invZ;

    // 3) radix select the KSEL-th largest score (monotone with soft_mask)
    for (int pass = 0; pass < 4; pass++) {
        int shift = 24 - pass * 8;
        hist[tid] = 0u; __syncthreads();
        unsigned pref = sh_prefix;
        unsigned hmask = (pass == 0) ? 0u : (0xFFFFFFFFu << (shift + 8));
        for (int i = tid; i < EPB; i += 256) {
            unsigned u = f2key(s[i]);
            if ((u & hmask) == (pref & hmask))
                atomicAdd(&hist[(u >> shift) & 255u], 1u);
        }
        __syncthreads();
        if (tid == 0) {
            unsigned rem = sh_remaining;
            for (int d = 255; d >= 0; d--) {
                if (rem <= hist[d]) { sh_prefix = pref | ((unsigned)d << shift); break; }
                rem -= hist[d];
            }
            sh_remaining = rem;
        }
        __syncthreads();
    }
    if (tid == 0) {
        unsigned u = sh_prefix;
        unsigned bits = (u & 0x80000000u) ? (u & 0x7FFFFFFFu) : ~u;
        sh_thr = __uint_as_float(bits);
    }
    __syncthreads();
    float thr = sh_thr;

    // 4) kept softmax mass
    float cs = 0.0f;
    for (int i = tid; i < EPB; i += 256) {
        float v = s[i];
        if (v >= thr) cs += __expf((v - mx) * INV_TEMP);
    }
    red[tid] = cs; __syncthreads();
    for (int w = 128; w > 0; w >>= 1) {
        if (tid < w) red[tid] += red[tid + w];
        __syncthreads();
    }
    if (tid == 0) {
        float mass = red[0] * invZ;
        stats[b]      = mx;
        stats[8 + b]  = invZ;
        stats[16 + b] = thr;
        stats[24 + b] = 1.0f / (mass + 1e-12f);
    }
}

// ---------------------------------------------------------------- finalize
__global__ __launch_bounds__(256)
void finalize_kernel(const float* __restrict__ edge, const float* __restrict__ stats,
                     float* __restrict__ causal, float* __restrict__ conf,
                     float* __restrict__ soft) {
    int e = blockIdx.x * 256 + threadIdx.x;   // 2048 blocks cover OUTR
    int b = e >> 16;
    float sEdge = edge[e];
    float sm = __expf((sEdge - stats[b]) * INV_TEMP) * stats[8 + b];
    float cm = (sEdge >= stats[16 + b]) ? sm : 0.0f;
    float ca = cm * stats[24 + b];
    causal[e] = ca;
    conf[e]   = 1.0f - ca;
    soft[e]   = sm;
}

// ---------------------------------------------------------------- launcher
extern "C" void kernel_launch(void* const* d_in, const int* in_sizes, int n_in,
                              void* d_out, int out_size, void* d_ws, size_t ws_size,
                              hipStream_t stream) {
    const float* x  = (const float*)d_in[0];
    const float* W1 = (const float*)d_in[1];
    const float* b1 = (const float*)d_in[2];
    const float* W2 = (const float*)d_in[3];
    const float* b2 = (const float*)d_in[4];
    float* out = (float*)d_out;

    char* ws = (char*)d_ws;
    _Float16* Xh = (_Float16*)ws;                        // 4 MB
    _Float16* Wh = (_Float16*)(ws + (4u << 20));         // 1 MB
    float*    P  = (float*)(ws + (5u << 20));            // 4 MB
    float* stats = (float*)(ws + (9u << 20));            // 128 B

    float* causal = out;
    float* conf   = out + OUTR;
    float* edge   = out + 2 * OUTR;   // edge_scores output region, used in-place
    float* soft   = out + 3 * OUTR;

    cvt_x_kernel   <<<2048, 256, 0, stream>>>(x, Xh);
    cvt_w1_kernel  <<<512,  256, 0, stream>>>(W1, Wh);
    gemm_kernel    <<<128,  256, 0, stream>>>(Xh, Wh, b1, P);
    edge_kernel    <<<128,  256, 0, stream>>>(P, W2, b2, edge);
    stats_kernel   <<<8,    256, 0, stream>>>(edge, stats);
    finalize_kernel<<<2048, 256, 0, stream>>>(edge, stats, causal, conf, soft);
}
